// UniversalCalculator_23579370455435
// MI455X (gfx1250) — matrix operand order repaired and checked by reading, not compiled, not run on hardware
//
#include <hip/hip_runtime.h>
#include <hip/hip_bf16.h>
#include <cstdint>
#include <cstddef>

// ---------------------------------------------------------------------------
// MoE forward (UniversalCalculator): y[t] = sum_k score[t,k] *
//      ( silu(x[t] @ W1[e] + b1[e]) @ W2[e] + b2[e] ),  e = topK_indices[t,k]
// Sparse dispatch (only routed experts computed), bf16 WMMA with f32 accum.
// ---------------------------------------------------------------------------

#define NUM_EXPERTS 8
#define TOKENS      8192
#define D_MODEL     1024
#define D_FF        1024
#define TOP_K       2
#define NASSIGN     (TOKENS * TOP_K)   // 16384

#define BM 128
#define BN 128
#define BK 32
#define LDA 40    // bf16 stride of A tile rows (80 B: 16B-aligned frags, no bank conflicts)
#define LDBU 20   // uint (k-pair) stride of B tile rows [n][kp] (80 B)

typedef __attribute__((ext_vector_type(16))) __bf16       v16bf;
typedef __attribute__((ext_vector_type(8)))  float        v8f;
typedef __attribute__((ext_vector_type(8)))  unsigned int v8u;
typedef __attribute__((ext_vector_type(4)))  unsigned int v4u;

__device__ __forceinline__ unsigned short bf16_bits(float f) {
  __hip_bfloat16 b = __float2bfloat16(f);
  return __builtin_bit_cast(unsigned short, b);
}
__device__ __forceinline__ unsigned int pack_bf16(float lo, float hi) {
  // v_cvt_pk_bf16_f32 path; __hip_bfloat162 is not trivially copyable -> memcpy
  __hip_bfloat162 b2 = __float22bfloat162_rn(float2{lo, hi});
  unsigned int u;
  __builtin_memcpy(&u, &b2, sizeof(u));
  return u;
}
__device__ __forceinline__ float fast_silu(float t) {
  // t * sigmoid(t) with v_rcp_f32 (no IEEE divide chain)
  return t * __builtin_amdgcn_rcpf(1.0f + __expf(-t));
}

// ---------------------------------------------------------------------------
// Kernel 1: routing. Bucket the 16384 (token,k) assignments by expert.
// ---------------------------------------------------------------------------
__global__ __launch_bounds__(256)
void moe_route(const int* __restrict__ topk_idx,
               const float* __restrict__ topk_scores,
               int* __restrict__ counts, int* __restrict__ offsets,
               int* __restrict__ tok_list, float* __restrict__ score_list,
               float* __restrict__ dropped_out) {
  __shared__ int s_cnt[NUM_EXPERTS];
  __shared__ int s_off[NUM_EXPERTS];
  __shared__ int s_cur[NUM_EXPERTS];
  const int tid = threadIdx.x;
  if (tid < NUM_EXPERTS) s_cnt[tid] = 0;
  __syncthreads();
  for (int i = tid; i < NASSIGN; i += 256) atomicAdd(&s_cnt[topk_idx[i]], 1);
  __syncthreads();
  if (tid == 0) {
    int acc = 0;
    for (int e = 0; e < NUM_EXPERTS; ++e) {
      s_off[e] = acc; s_cur[e] = acc; acc += s_cnt[e];
    }
    *dropped_out = -1.0f;   // reference's num_dropped_tokens output
  }
  __syncthreads();
  if (tid < NUM_EXPERTS) { counts[tid] = s_cnt[tid]; offsets[tid] = s_off[tid]; }
  for (int i = tid; i < NASSIGN; i += 256) {
    int e = topk_idx[i];
    int pos = atomicAdd(&s_cur[e], 1);
    tok_list[pos]   = i >> 1;          // TOP_K == 2
    score_list[pos] = topk_scores[i];  // SCORE_SCALE == 1.0
  }
}

// ---------------------------------------------------------------------------
// Kernel 2/3: tiled expert GEMM.  FIRST: H = silu(gather(x) @ W1 + b1) (bf16)
//             !FIRST: y[tok] += score * (H @ W2 + b2)   (f32 atomics)
// Block: 256 threads = 8 wave32s in a 4(M)x2(N) grid; each wave owns a
// 32x64 patch = 2x4 v_wmma_f32_16x16x32_bf16 tiles.  K-tile = 32.
// LDS layouts chosen so every WMMA fragment is 2x contiguous ds_load_b128.
// ---------------------------------------------------------------------------
template <bool FIRST>
__global__ __launch_bounds__(256)
void moe_gemm(const float* __restrict__ Xf,            // FIRST: x [TOKENS][1024]
              const unsigned short* __restrict__ Ha,   // !FIRST: H bf16 [NASSIGN][1024]
              const float* __restrict__ W,             // W1/W2 [8][1024][1024]
              const float* __restrict__ bias,          // b1/b2 [8][1024]
              const int* __restrict__ counts,
              const int* __restrict__ offsets,
              const int* __restrict__ tok_list,
              const float* __restrict__ score_list,
              unsigned short* __restrict__ H_out,      // FIRST
              float* __restrict__ Y) {                 // !FIRST
  __shared__ unsigned short lsA[BM * LDA];   // bf16 bits, [m][k] row-major, 10 KB
  __shared__ unsigned int   lsB[BN * LDBU];  // [n][kp] packed (2k,2k+1) pairs, 10 KB

  const int e     = blockIdx.z;
  const int count = counts[e];
  const int m0    = blockIdx.x * BM;
  if (m0 >= count) return;                             // block-uniform exit
  const int off = offsets[e];
  const int n0  = blockIdx.y * BN;

  const int tid  = threadIdx.x;
  const int lane = tid & 31;
  const int wave = tid >> 5;
  const int h    = lane >> 4;         // lane half (0/1)
  const int ln   = lane & 15;
  const int wm   = (wave >> 1) * 32;  // wave M offset in block
  const int wn   = (wave & 1) * 64;   // wave N offset in block

  // Gather source rows for the A staging (fixed across K loop).
  int rowA[4];
#pragma unroll
  for (int q = 0; q < 4; ++q) {
    int m  = (tid + q * 256) >> 3;
    int ar = m0 + m;
    if (FIRST) rowA[q] = (ar < count) ? tok_list[off + ar] : 0;
    else       rowA[q] = (ar < count) ? (off + ar) : 0;  // H is pre-gathered
  }

  const v8f vzero = {0.f, 0.f, 0.f, 0.f, 0.f, 0.f, 0.f, 0.f};
  v8f acc[2][4];
#pragma unroll
  for (int i = 0; i < 2; ++i)
#pragma unroll
    for (int j = 0; j < 4; ++j) acc[i][j] = vzero;

  const float* Wb = W + ((size_t)e << 20);

  for (int k0 = 0; k0 < 1024; k0 += BK) {
    // ---- stage A tile (BM x BK) into LDS as bf16 ----
#pragma unroll
    for (int q = 0; q < 4; ++q) {
      int linear = tid + q * 256;
      int m  = linear >> 3;
      int kq = (linear & 7) * 4;
      unsigned int p0, p1;
      if (FIRST) {
        const float4 v = *(const float4*)(Xf + (size_t)rowA[q] * D_MODEL + k0 + kq);
        p0 = pack_bf16(v.x, v.y);
        p1 = pack_bf16(v.z, v.w);
      } else {
        const uint2 v = *(const uint2*)(Ha + (size_t)rowA[q] * D_FF + k0 + kq);
        p0 = v.x; p1 = v.y;
      }
      *(uint2*)&lsA[m * LDA + kq] = make_uint2(p0, p1);
    }
    // ---- stage B tile (BK x BN) as [n][kpair] so frags are contiguous ----
#pragma unroll
    for (int q = 0; q < 4; ++q) {
      int linear = tid + q * 256;
      int k  = linear >> 5;           // 0..31
      int n4 = (linear & 31) * 4;     // 0..124
      const float4 v = *(const float4*)(Wb + (size_t)(k0 + k) * 1024 + n0 + n4);
      unsigned short* bp = (unsigned short*)lsB;
      int kp = k >> 1, par = k & 1;
      bp[(((n4 + 0) * LDBU + kp) << 1) + par] = bf16_bits(v.x);
      bp[(((n4 + 1) * LDBU + kp) << 1) + par] = bf16_bits(v.y);
      bp[(((n4 + 2) * LDBU + kp) << 1) + par] = bf16_bits(v.z);
      bp[(((n4 + 3) * LDBU + kp) << 1) + par] = bf16_bits(v.w);
    }
    // prefetch next weight k-tile (streams through L2) -> global_prefetch_b8
    if (k0 + BK < 1024) {
      const float* nxt = Wb + (size_t)(k0 + BK + (tid >> 3)) * 1024 + n0 + (tid & 7) * 16;
      __builtin_prefetch(nxt, 0, 0);
    }
    __syncthreads();

    // ---- fragments: 2x ds_load_b128 each (ISA 16-bit A/B lane layouts) ----
    v16bf afr[2], bfr[4];
#pragma unroll
    for (int sm = 0; sm < 2; ++sm) {
      int m = wm + sm * 16 + ln;
      // VGPR 0..3: k = 8h..8h+7 ; VGPR 4..7: k = 16+8h..23+8h  (contiguous runs)
      const v4u lo = *(const v4u*)__builtin_assume_aligned(&lsA[m * LDA + 8 * h], 16);
      const v4u hi = *(const v4u*)__builtin_assume_aligned(&lsA[m * LDA + 16 + 8 * h], 16);
      afr[sm] = __builtin_bit_cast(v16bf, __builtin_shufflevector(lo, hi, 0, 1, 2, 3, 4, 5, 6, 7));
    }
#pragma unroll
    for (int sn = 0; sn < 4; ++sn) {
      int n = wn + sn * 16 + ln;
      // VGPR 0..3: kpair = 4h..4h+3 ; VGPR 4..7: kpair = 8+4h..11+4h (matches A)
      const v4u lo = *(const v4u*)__builtin_assume_aligned(&lsB[n * LDBU + 4 * h], 16);
      const v4u hi = *(const v4u*)__builtin_assume_aligned(&lsB[n * LDBU + 8 + 4 * h], 16);
      bfr[sn] = __builtin_bit_cast(v16bf, __builtin_shufflevector(lo, hi, 0, 1, 2, 3, 4, 5, 6, 7));
    }

#pragma unroll
    for (int sm = 0; sm < 2; ++sm)
#pragma unroll
      for (int sn = 0; sn < 4; ++sn)
        acc[sm][sn] = __builtin_amdgcn_wmma_f32_16x16x32_bf16(
            false, afr[sm], false, bfr[sn], (short)0, acc[sm][sn], false, false);
    __syncthreads();
  }

  // ---- epilogue: hoist per-row work out of the sn loop ----
  float bv[4];
#pragma unroll
  for (int sn = 0; sn < 4; ++sn) bv[sn] = bias[(e << 10) + n0 + wn + sn * 16 + ln];

#pragma unroll
  for (int sm = 0; sm < 2; ++sm) {
#pragma unroll
    for (int v = 0; v < 8; ++v) {
      int ar = m0 + wm + sm * 16 + h * 8 + v;   // C layout: M = v + 8*half
      if (ar < count) {
        if (FIRST) {
          unsigned short* hrow = H_out + (size_t)(off + ar) * D_FF + n0 + wn + ln;
#pragma unroll
          for (int sn = 0; sn < 4; ++sn)
            hrow[sn * 16] = bf16_bits(fast_silu(acc[sm][sn][v] + bv[sn]));
        } else {
          int pos  = off + ar;
          int tok  = tok_list[pos];
          float sc = score_list[pos];
          float* yrow = Y + (size_t)tok * D_MODEL + n0 + wn + ln;
#pragma unroll
          for (int sn = 0; sn < 4; ++sn)
            atomicAdd(yrow + sn * 16, (acc[sm][sn][v] + bv[sn]) * sc);
        }
      }
    }
  }
}

// ---------------------------------------------------------------------------
// Host launcher.  Workspace layout (needs ~33.9 MB):
//   [0)       counts[8] int, offsets[8] int
//   [1 KB)    tok_list[16384] int
//   [66.5 KB) score_list[16384] float
//   [256 KB)  H bf16 [16384][1024]  (32 MB)
// ---------------------------------------------------------------------------
extern "C" void kernel_launch(void* const* d_in, const int* in_sizes, int n_in,
                              void* d_out, int out_size, void* d_ws, size_t ws_size,
                              hipStream_t stream) {
  const float* x           = (const float*)d_in[0];
  const float* topk_scores = (const float*)d_in[1];
  const float* W1          = (const float*)d_in[2];
  const float* b1          = (const float*)d_in[3];
  const float* W2          = (const float*)d_in[4];
  const float* b2          = (const float*)d_in[5];
  const int*   topk_idx    = (const int*)d_in[6];

  float* y       = (float*)d_out;
  float* dropped = y + (size_t)TOKENS * D_MODEL;  // second tuple output

  char* ws = (char*)d_ws;
  int*            counts     = (int*)ws;
  int*            offsets    = counts + 8;
  int*            tok_list   = (int*)(ws + 1024);
  float*          score_list = (float*)(ws + 1024 + NASSIGN * sizeof(int));
  unsigned short* H          = (unsigned short*)(ws + 256 * 1024);

  // y accumulated via atomics -> zero it (memset node is graph-capture safe)
  (void)hipMemsetAsync(d_out, 0, (size_t)TOKENS * D_MODEL * sizeof(float), stream);

  moe_route<<<1, 256, 0, stream>>>(topk_idx, topk_scores, counts, offsets,
                                   tok_list, score_list, dropped);

  dim3 grid(NASSIGN / BM, D_FF / BN, NUM_EXPERTS);  // (128, 8, 8); early-exit on counts
  moe_gemm<true><<<grid, 256, 0, stream>>>(x, nullptr, W1, b1, counts, offsets,
                                           tok_list, nullptr, H, nullptr);
  moe_gemm<false><<<grid, 256, 0, stream>>>(nullptr, H, W2, b2, counts, offsets,
                                            tok_list, score_list, nullptr, y);
}
